// GVPConvLayer_12034498363922
// MI455X (gfx1250) — matrix operand order repaired and checked by reading, not compiled, not run on hardware
//
#include <hip/hip_runtime.h>
#include <hip/hip_bf16.h>
#include <math.h>

// ---------------------------------------------------------------------------
// GVP conv layer for MI455X (gfx1250, wave32, WMMA)
// Edge kernel: 32 edges/block, 8 waves; each wave owns one 16-col N-tile and
// two 16-row M-tiles -> B fragment reused for 2 WMMAs per K-step.
// ---------------------------------------------------------------------------

#define N_NODES   50000
#define N_EDGES   800000
#define NS        128
#define NV        16
#define ES        32
#define MSG_SI    288          // 2*NS + ES
#define MSG_VI    33           // 2*NV + 1
#define MSG_H     33
#define MSG_K     321          // MSG_SI + MSG_H
#define KPAD_MSG  352          // 321 padded to 11*32
#define UPD_K     144          // NS + NV
#define KPAD_UPD  160          // 144 padded to 5*32
#define LN_EPS    1e-5f
#define E_PER_BLK 32

typedef __attribute__((ext_vector_type(16))) __bf16 v16bf;
typedef __attribute__((ext_vector_type(8)))  __bf16 v8bf;
typedef __attribute__((ext_vector_type(8)))  float  v8f;

// Load a 16-bit A/B operand fragment as two contiguous 16B chunks.
__device__ __forceinline__ v16bf frag_from_chunks(const __bf16* p, int second_off) {
    v8bf lo = *(const v8bf*)p;
    v8bf hi = *(const v8bf*)(p + second_off);
    return __builtin_shufflevector(lo, hi, 0,1,2,3,4,5,6,7,8,9,10,11,12,13,14,15);
}

// ---------------------------------------------------------------------------
// 0) zero accumulators
// ---------------------------------------------------------------------------
__global__ __launch_bounds__(256)
void gvp_zero_kernel(float* __restrict__ p, int n) {
    int i = blockIdx.x * 256 + threadIdx.x;
    if (i < n) p[i] = 0.0f;
}

// ---------------------------------------------------------------------------
// 1) repack Ws weights to bf16, K-padded (row-major [out=128][Kpad])
// ---------------------------------------------------------------------------
__global__ __launch_bounds__(256)
void gvp_prep_weights(const float* __restrict__ Wm,   // [128][321]
                      const float* __restrict__ Wu,   // [128][144]
                      __bf16* __restrict__ wmsg,      // [128][352]
                      __bf16* __restrict__ wupd) {    // [128][160]
    int i = blockIdx.x * 256 + threadIdx.x;
    if (i < 128 * KPAD_MSG) {
        int n = i / KPAD_MSG, k = i % KPAD_MSG;
        wmsg[i] = (k < MSG_K) ? (__bf16)Wm[n * MSG_K + k] : (__bf16)0.0f;
    } else {
        int j = i - 128 * KPAD_MSG;
        if (j < 128 * KPAD_UPD) {
            int n = j / KPAD_UPD, k = j % KPAD_UPD;
            wupd[j] = (k < UPD_K) ? (__bf16)Wu[n * UPD_K + k] : (__bf16)0.0f;
        }
    }
}

// ---------------------------------------------------------------------------
// 2) edge kernel: message GVP + atomic scatter-add aggregation
// ---------------------------------------------------------------------------
__global__ __launch_bounds__(256)
void gvp_msg_kernel(const float* __restrict__ node_s, const float* __restrict__ node_v,
                    const float* __restrict__ edge_s, const float* __restrict__ edge_v,
                    const int*   __restrict__ eidx,
                    const float* __restrict__ Wv,   // [33][33]
                    const float* __restrict__ Wv2,  // [16][33]
                    const float* __restrict__ Ws_b, // [128]
                    const __bf16* __restrict__ Wmsg,// [128][352] bf16
                    float* __restrict__ agg_s, float* __restrict__ agg_v,
                    float* __restrict__ cnt) {
    __shared__ int   sh_src[E_PER_BLK], sh_dst[E_PER_BLK];
    __shared__ float sh_mv[E_PER_BLK][100];   // msg_v  [edge][33*3]
    __shared__ float sh_vh[E_PER_BLK][100];   // v_h    [edge][33*3]
    __shared__ __align__(16) __bf16 sh_sin[E_PER_BLK][KPAD_MSG];
    __shared__ float sh_Wv[MSG_H * MSG_VI];   // 1089
    __shared__ float sh_Wv2[NV * MSG_H];      // 528

    const int tid = threadIdx.x;
    const int e   = tid >> 3;     // edge slot 0..31
    const int s   = tid & 7;      // sub-lane within edge (8 threads/edge)
    const int ebase = blockIdx.x * E_PER_BLK;

    if (tid < E_PER_BLK) {
        sh_src[tid] = eidx[ebase + tid];
        sh_dst[tid] = eidx[N_EDGES + ebase + tid];
    }
    for (int i = tid; i < MSG_H * MSG_VI; i += 256) sh_Wv[i]  = Wv[i];
    for (int i = tid; i < NV * MSG_H;    i += 256) sh_Wv2[i] = Wv2[i];
    __syncthreads();

    const int src = sh_src[e], dst = sh_dst[e];
    const int ge  = ebase + e;

    // gather msg_v = concat(node_v[src], edge_v, node_v[dst])  (33 x 3 f32)
    for (int j = s; j < 99; j += 8) {
        float val;
        if (j < 48)      val = node_v[src * 48 + j];
        else if (j < 51) val = edge_v[ge * 3 + (j - 48)];
        else             val = node_v[dst * 48 + (j - 51)];
        sh_mv[e][j] = val;
    }
    // gather scalar part of s_in (288) -> bf16 LDS, and zero tail pad
    for (int j = s; j < MSG_SI; j += 8) {
        float val;
        if (j < 128)      val = node_s[src * 128 + j];
        else if (j < 160) val = edge_s[ge * 32 + (j - 128)];
        else              val = node_s[dst * 128 + (j - 160)];
        sh_sin[e][j] = (__bf16)val;
    }
    for (int j = MSG_K + s; j < KPAD_MSG; j += 8) sh_sin[e][j] = (__bf16)0.0f;
    __syncthreads();

    // v_h = einsum('vi,hv->hi'): [33][3]; norms -> s_in[288..320]
    for (int h = s; h < MSG_H; h += 8) {
        float a0 = 0.f, a1 = 0.f, a2 = 0.f;
        for (int v = 0; v < MSG_VI; ++v) {
            float w = sh_Wv[h * MSG_VI + v];
            a0 += w * sh_mv[e][v * 3 + 0];
            a1 += w * sh_mv[e][v * 3 + 1];
            a2 += w * sh_mv[e][v * 3 + 2];
        }
        sh_vh[e][h * 3 + 0] = a0;
        sh_vh[e][h * 3 + 1] = a1;
        sh_vh[e][h * 3 + 2] = a2;
        sh_sin[e][MSG_SI + h] = (__bf16)sqrtf(a0 * a0 + a1 * a1 + a2 * a2);
    }
    __syncthreads();

    // v_out = Wv2 @ v_h (16 x 3), sigmoid-norm gate, atomic scatter agg_v
    for (int o = s; o < NV; o += 8) {
        float a0 = 0.f, a1 = 0.f, a2 = 0.f;
        for (int h = 0; h < MSG_H; ++h) {
            float w = sh_Wv2[o * MSG_H + h];
            a0 += w * sh_vh[e][h * 3 + 0];
            a1 += w * sh_vh[e][h * 3 + 1];
            a2 += w * sh_vh[e][h * 3 + 2];
        }
        float nrm  = sqrtf(a0 * a0 + a1 * a1 + a2 * a2);
        float gate = 1.0f / (1.0f + __expf(-nrm));
        atomicAdd(&agg_v[dst * 48 + o * 3 + 0], a0 * gate);
        atomicAdd(&agg_v[dst * 48 + o * 3 + 1], a1 * gate);
        atomicAdd(&agg_v[dst * 48 + o * 3 + 2], a2 * gate);
    }
    if (s == 0) atomicAdd(&cnt[dst], 1.0f);

    // WMMA phase: wave -> N-tile; two M-tiles share each B fragment
    const int wave = tid >> 5;       // n-tile 0..7
    const int lane = tid & 31;
    const int half = lane >> 4;
    const int lm   = lane & 15;

    v8f acc0 = {}, acc1 = {};
    for (int ks = 0; ks < KPAD_MSG / 32; ++ks) {
        // B: 32x16 bf16; lane holds col (wave*16+lm), K vals in table order (8h+s, 16+8h+s)
        v16bf B  = frag_from_chunks(&Wmsg[(wave * 16 + lm) * KPAD_MSG + ks * 32 + half * 8], 16);
        // A: 16x32 bf16; lane holds row lm of each M-tile
        v16bf A0 = frag_from_chunks(&sh_sin[lm][ks * 32 + half * 8], 16);
        v16bf A1 = frag_from_chunks(&sh_sin[16 + lm][ks * 32 + half * 8], 16);
        acc0 = __builtin_amdgcn_wmma_f32_16x16x32_bf16(false, A0, false, B,
                                                       (short)0, acc0, false, false);
        acc1 = __builtin_amdgcn_wmma_f32_16x16x32_bf16(false, A1, false, B,
                                                       (short)0, acc1, false, false);
    }
    const int   n    = wave * 16 + lm;
    const float bias = Ws_b[n];
    for (int r = 0; r < 8; ++r) {
        int   m = r + half * 8;                    // edge slot, M-tile 0
        float v = acc0[r] + bias;
        v = v > 0.f ? v : 0.f;                     // relu BEFORE aggregation
        atomicAdd(&agg_s[sh_dst[m] * 128 + n], v);
    }
    for (int r = 0; r < 8; ++r) {
        int   m = 16 + r + half * 8;               // edge slot, M-tile 1
        float v = acc1[r] + bias;
        v = v > 0.f ? v : 0.f;
        atomicAdd(&agg_s[sh_dst[m] * 128 + n], v);
    }
}

// ---------------------------------------------------------------------------
// 3) node kernel: mean-normalize, update GVP (WMMA for 144->128), residual,
//    LayerNorm; writes both outputs. block = 16 nodes, 256 threads.
// ---------------------------------------------------------------------------
__global__ __launch_bounds__(256)
void gvp_upd_kernel(const float* __restrict__ node_s, const float* __restrict__ node_v,
                    const float* __restrict__ Wv,   // [16][16]
                    const float* __restrict__ Wv2,  // [16][16]
                    const float* __restrict__ Ws_b, // [128]
                    const __bf16* __restrict__ Wupd,// [128][160] bf16
                    const float* __restrict__ ln_g, const float* __restrict__ ln_b,
                    const float* __restrict__ agg_s, const float* __restrict__ agg_v,
                    const float* __restrict__ cnt,
                    float* __restrict__ out_s, float* __restrict__ out_v) {
    __shared__ __align__(16) __bf16 sh_sin[16][KPAD_UPD];
    __shared__ float sh_av[16][50];
    __shared__ float sh_vh[16][50];
    __shared__ float sh_Wv[256], sh_Wv2[256];
    __shared__ float sh_inv[16];
    __shared__ float sh_delta[16][128];

    const int tid = threadIdx.x;
    const int e   = tid >> 4;
    const int s   = tid & 15;
    const int nbase = blockIdx.x * 16;
    const int node  = nbase + e;

    if (tid < 16) sh_inv[tid] = 1.0f / fmaxf(cnt[nbase + tid], 1.0f);
    for (int i = tid; i < 256; i += 256) { sh_Wv[i] = Wv[i]; sh_Wv2[i] = Wv2[i]; }
    __syncthreads();

    const float inv = sh_inv[e];
    for (int j = s; j < 128; j += 16)
        sh_sin[e][j] = (__bf16)(agg_s[node * 128 + j] * inv);
    for (int j = s; j < 48; j += 16)
        sh_av[e][j] = agg_v[node * 48 + j] * inv;
    sh_sin[e][UPD_K + s] = (__bf16)0.0f;   // pad 144..159
    __syncthreads();

    // v_h (16 x 3) + norms -> s_in[128..143]
    {
        const int h = s;
        float a0 = 0.f, a1 = 0.f, a2 = 0.f;
        for (int v = 0; v < NV; ++v) {
            float w = sh_Wv[h * NV + v];
            a0 += w * sh_av[e][v * 3 + 0];
            a1 += w * sh_av[e][v * 3 + 1];
            a2 += w * sh_av[e][v * 3 + 2];
        }
        sh_vh[e][h * 3 + 0] = a0;
        sh_vh[e][h * 3 + 1] = a1;
        sh_vh[e][h * 3 + 2] = a2;
        sh_sin[e][NS + h] = (__bf16)sqrtf(a0 * a0 + a1 * a1 + a2 * a2);
    }
    __syncthreads();

    // v_out + gate + residual -> node_v_out
    {
        const int o = s;
        float a0 = 0.f, a1 = 0.f, a2 = 0.f;
        for (int h = 0; h < NV; ++h) {
            float w = sh_Wv2[o * NV + h];
            a0 += w * sh_vh[e][h * 3 + 0];
            a1 += w * sh_vh[e][h * 3 + 1];
            a2 += w * sh_vh[e][h * 3 + 2];
        }
        float nrm  = sqrtf(a0 * a0 + a1 * a1 + a2 * a2);
        float gate = 1.0f / (1.0f + __expf(-nrm));
        out_v[node * 48 + o * 3 + 0] = node_v[node * 48 + o * 3 + 0] + a0 * gate;
        out_v[node * 48 + o * 3 + 1] = node_v[node * 48 + o * 3 + 1] + a1 * gate;
        out_v[node * 48 + o * 3 + 2] = node_v[node * 48 + o * 3 + 2] + a2 * gate;
    }

    // WMMA: delta_s = relu(s_in @ Wupd^T + b)
    const int wave = tid >> 5;
    const int lane = tid & 31;
    const int half = lane >> 4;
    const int lm   = lane & 15;

    v8f acc = {};
    for (int ks = 0; ks < KPAD_UPD / 32; ++ks) {
        v16bf A = frag_from_chunks(&sh_sin[lm][ks * 32 + half * 8], 16);
        v16bf B = frag_from_chunks(&Wupd[(wave * 16 + lm) * KPAD_UPD + ks * 32 + half * 8], 16);
        acc = __builtin_amdgcn_wmma_f32_16x16x32_bf16(false, A, false, B,
                                                      (short)0, acc, false, false);
    }
    const int   n    = wave * 16 + lm;
    const float bias = Ws_b[n];
    for (int r = 0; r < 8; ++r) {
        int   m = r + half * 8;
        float v = acc[r] + bias;
        sh_delta[m][n] = v > 0.f ? v : 0.f;
    }
    __syncthreads();

    // residual + LayerNorm over 128 channels; half-wave shfl reduction
    float x[8];
    float sum = 0.f, ss = 0.f;
    const int base = s * 8;
    for (int k = 0; k < 8; ++k) {
        float xv = node_s[node * 128 + base + k] + sh_delta[e][base + k];
        x[k] = xv; sum += xv; ss += xv * xv;
    }
    for (int off = 1; off < 16; off <<= 1) {
        sum += __shfl_xor(sum, off, 16);
        ss  += __shfl_xor(ss,  off, 16);
    }
    const float mu   = sum * (1.0f / 128.0f);
    const float var  = ss * (1.0f / 128.0f) - mu * mu;
    const float rstd = rsqrtf(var + LN_EPS);
    for (int k = 0; k < 8; ++k)
        out_s[node * 128 + base + k] =
            (x[k] - mu) * rstd * ln_g[base + k] + ln_b[base + k];
}

// ---------------------------------------------------------------------------
// launch
// ---------------------------------------------------------------------------
extern "C" void kernel_launch(void* const* d_in, const int* in_sizes, int n_in,
                              void* d_out, int out_size, void* d_ws, size_t ws_size,
                              hipStream_t stream) {
    (void)in_sizes; (void)n_in; (void)out_size; (void)ws_size;

    const float* node_s   = (const float*)d_in[0];
    const float* node_v   = (const float*)d_in[1];
    const float* edge_s   = (const float*)d_in[2];
    const float* edge_v   = (const float*)d_in[3];
    const int*   eidx     = (const int*)  d_in[4];
    const float* msg_Wv   = (const float*)d_in[5];
    const float* msg_Wv2  = (const float*)d_in[6];
    const float* msg_Ws_w = (const float*)d_in[7];
    const float* msg_Ws_b = (const float*)d_in[8];
    const float* upd_Wv   = (const float*)d_in[9];
    const float* upd_Wv2  = (const float*)d_in[10];
    const float* upd_Ws_w = (const float*)d_in[11];
    const float* upd_Ws_b = (const float*)d_in[12];
    const float* ln_g     = (const float*)d_in[13];
    const float* ln_b     = (const float*)d_in[14];

    // workspace layout (f32 accumulators then bf16 weights; 16B aligned)
    float*  agg_s = (float*)d_ws;                       // 50000*128
    float*  agg_v = agg_s + (size_t)N_NODES * 128;      // 50000*48
    float*  cntp  = agg_v + (size_t)N_NODES * 48;       // 50000
    __bf16* wmsg  = (__bf16*)(cntp + N_NODES);          // 128*352
    __bf16* wupd  = wmsg + 128 * KPAD_MSG;              // 128*160

    float* out_s = (float*)d_out;
    float* out_v = out_s + (size_t)N_NODES * 128;

    const int n_zero = N_NODES * 128 + N_NODES * 48 + N_NODES; // 8,850,000
    gvp_zero_kernel<<<(n_zero + 255) / 256, 256, 0, stream>>>(agg_s, n_zero);

    const int n_w = 128 * KPAD_MSG + 128 * KPAD_UPD;           // 65,536
    gvp_prep_weights<<<(n_w + 255) / 256, 256, 0, stream>>>(msg_Ws_w, upd_Ws_w,
                                                            wmsg, wupd);

    gvp_msg_kernel<<<N_EDGES / E_PER_BLK, 256, 0, stream>>>(
        node_s, node_v, edge_s, edge_v, eidx,
        msg_Wv, msg_Wv2, msg_Ws_b, wmsg,
        agg_s, agg_v, cntp);

    gvp_upd_kernel<<<N_NODES / 16, 256, 0, stream>>>(
        node_s, node_v,
        upd_Wv, upd_Wv2, upd_Ws_b, wupd,
        ln_g, ln_b,
        agg_s, agg_v, cntp,
        out_s, out_v);
}
